// Attention_10015863734951
// MI455X (gfx1250) — matrix operand order repaired and checked by reading, not compiled, not run on hardware
//
#include <hip/hip_runtime.h>
#include <cstdint>
#include <cstddef>

// ---------------------------------------------------------------------------
// Bahdanau attention, restructured for MI455X (gfx1250, wave32, WMMA).
//   B=16, S=8192, D=512, H=1024
// Key identity: cat = [enc | hidden] =>
//   y[b,s,o] = enc[b,s,:] @ W[o,:512] + (hidden[b,:] @ W[o,512:] + b_attn[o])
// "hbias" (s-independent) is folded into the WMMA accumulator init.
// Main pass: fused bf16 WMMA GEMM (M=131072,K=512,N=1024); the [B,S,H]
// energy tensor is never materialized. W[:, :512] is pre-packed once into
// bf16 in the exact WMMA B-lane layout (1 MB, L2-resident): zero B-side
// converts, contiguous 1KB-per-wave B tile loads.
// Inner loop keeps minimal register pressure (B consumed immediately);
// L2 load latency is hidden by wave-level parallelism (8 waves/WG, ~120
// VGPRs -> deep SIMD residency), which costs zero extra issue slots,
// unlike register-staged pipelining (which forced v_mov juggling).
// tanh uses the hardware transcendental path (v_exp_f32 + v_rcp_f32).
// ---------------------------------------------------------------------------

typedef __attribute__((ext_vector_type(4)))  float  v4f;
typedef __attribute__((ext_vector_type(8)))  float  v8f;
typedef __attribute__((ext_vector_type(16))) float  v16f;
typedef __attribute__((ext_vector_type(16))) __bf16 v16bf;

#define S_LEN 8192
#define D_DIM 512
#define H_DIM 1024
#define B_DIM 16
#define NTILES 64           // N / 16
#define KSTEPS 16           // K / 32
#define CTX_CH 16           // context chunks per batch (512 rows each)

// tanh(x) = 1 - 2/(exp(2x)+1); v_exp_f32 + v_rcp_f32, ~5 ops vs ~20 for libm.
__device__ inline float fast_tanh(float x) {
  float e = __expf(2.0f * x);
  return 1.0f - 2.0f * __builtin_amdgcn_rcpf(e + 1.0f);
}

// ---------------------------------------------------------------------------
// Kernel 0: hbias[b,o] = hidden[b,:] @ W[o, 512:1024] + b_attn[o]
// ---------------------------------------------------------------------------
__global__ __launch_bounds__(256) void hbias_kernel(
    const float* __restrict__ hidden, const float* __restrict__ W,
    const float* __restrict__ b_attn, float* __restrict__ hbias) {
  int idx = blockIdx.x * 256 + threadIdx.x;      // [0, 16384)
  int b = idx >> 10;
  int o = idx & (H_DIM - 1);
  const v4f* h4 = (const v4f*)(hidden + (size_t)b * D_DIM);
  const v4f* w4 = (const v4f*)(W + (size_t)o * H_DIM + D_DIM);
  float s = 0.0f;
#pragma unroll 4
  for (int i = 0; i < D_DIM / 4; ++i) {
    v4f a = h4[i], w = w4[i];
    s += a.x * w.x + a.y * w.y + a.z * w.z + a.w * w.w;
  }
  hbias[idx] = s + b_attn[o];
}

// ---------------------------------------------------------------------------
// Kernel 1: pack W[:, :512] -> bf16 in WMMA B-operand lane layout.
// Wp[((ks*64 + nt)*32 + lane)*16 + e] =
//     bf16( W[nt*16 + (lane&15)][ks*32 + (lane>>4)*16 + e] )
// ---------------------------------------------------------------------------
__global__ __launch_bounds__(256) void wpack_kernel(
    const float* __restrict__ W, __bf16* __restrict__ Wp) {
  int idx = blockIdx.x * 256 + threadIdx.x;   // [0, KSTEPS*NTILES*32) = 32768
  int lane = idx & 31;
  int tile = idx >> 5;                        // ks*64 + nt
  int nt = tile & (NTILES - 1);
  int ks = tile >> 6;
  int o = nt * 16 + (lane & 15);
  int k = ks * 32 + (lane >> 4) * 16;
  v16f x = *(const v16f*)(W + (size_t)o * H_DIM + k);   // 64B-aligned
  *(v16bf*)(Wp + (size_t)idx * 16) = __builtin_convertvector(x, v16bf);
}

// ---------------------------------------------------------------------------
// Kernel 2: fused WMMA score kernel.
// Grid: 8192 workgroups (one per 16-row M tile), 256 threads = 8 waves.
// Each wave: 16 rows x 128 cols (8 n-tiles), K loop 512 in steps of 32.
// A tile (enc) layout (contiguous k, same order as the Wp B packing):
//   lane<16: M=lane, elems {0..15}=K[0..15]
//   lane>=16: same M, elems {0..15}=K[16..31]
// B tiles come pre-packed in lane layout from Wp (bf16, no converts).
// hbias is pre-loaded into the accumulators (C/D lane column is constant
// across the 8 accumulator VGPRs).
// ---------------------------------------------------------------------------
__global__ __launch_bounds__(256) void score_kernel(
    const float* __restrict__ enc, const __bf16* __restrict__ Wp,
    const float* __restrict__ hbias, const float* __restrict__ vw,
    float* __restrict__ scores) {
  __shared__ float s_sc[16];
  const int tid    = threadIdx.x;
  const int wave   = tid >> 5;
  const int lane   = tid & 31;
  const int lane15 = lane & 15;
  const int half   = lane >> 4;
  if (tid < 16) s_sc[tid] = 0.0f;
  __syncthreads();

  const size_t row  = (size_t)blockIdx.x * 16 + lane15;  // flat row in [0, B*S)
  const float* Arow = enc + row * D_DIM + half * 16;     // A half-row base
  const int b = blockIdx.x >> 9;   // 512 blocks per batch (8192 rows / 16)

  // Init accumulators with hbias[b, o] (o fixed per lane per tile).
  v8f acc[8];
  float vvs[8];
#pragma unroll
  for (int t = 0; t < 8; ++t) {
    int o = wave * 128 + t * 16 + lane15;
    float h = hbias[b * H_DIM + o];
    vvs[t] = vw[o];
    acc[t] = (v8f){h, h, h, h, h, h, h, h};
  }

  for (int ks = 0; ks < KSTEPS; ++ks) {
    const int k0 = ks * 32;
    // --- A operand: two contiguous 8-float runs per lane, cvt -> bf16 ---
    v8f alo = *(const v8f*)(Arow + k0);
    v8f ahi = *(const v8f*)(Arow + k0 + 8);
    v16f af = __builtin_shufflevector(alo, ahi,
                 0, 1, 2, 3, 4, 5, 6, 7, 8, 9, 10, 11, 12, 13, 14, 15);
    v16bf a = __builtin_convertvector(af, v16bf);
    // --- B operands: contiguous pre-packed bf16 tiles ---
    const __bf16* bp =
        Wp + (((size_t)ks * NTILES + (size_t)wave * 8) * 32 + lane) * 16;
#pragma unroll
    for (int t = 0; t < 8; ++t) {
      v16bf bb = *(const v16bf*)(bp + (size_t)t * (32 * 16));
      acc[t] = __builtin_amdgcn_wmma_f32_16x16x32_bf16(
          false, a, false, bb, (short)0, acc[t], false, false);
    }
  }

  // Epilogue: scores[row] = sum_o tanh(acc) * v_w[o]
  // C/D layout: lane L, vgpr v -> M = v + 8*(L>>4), N(col) = L&15.
  float rsum[8] = {0, 0, 0, 0, 0, 0, 0, 0};
#pragma unroll
  for (int t = 0; t < 8; ++t) {
    float vv = vvs[t];
#pragma unroll
    for (int v = 0; v < 8; ++v)
      rsum[v] += fast_tanh(acc[t][v]) * vv;
  }
  // reduce across the 16 lanes that share each row (xor 1,2,4,8 stays in-half)
#pragma unroll
  for (int off = 8; off > 0; off >>= 1) {
#pragma unroll
    for (int v = 0; v < 8; ++v)
      rsum[v] += __shfl_xor(rsum[v], off, 32);
  }
  if (lane15 == 0) {
#pragma unroll
    for (int v = 0; v < 8; ++v)
      atomicAdd(&s_sc[half * 8 + v], rsum[v]);   // ds_add_f32, 2 lanes/wave
  }
  __syncthreads();
  if (tid < 16) scores[(size_t)blockIdx.x * 16 + tid] = s_sc[tid];
}

// ---------------------------------------------------------------------------
// Kernel 3: softmax over S per batch. One workgroup per b.
// ---------------------------------------------------------------------------
__global__ __launch_bounds__(256) void softmax_kernel(
    const float* __restrict__ scores, float* __restrict__ weights) {
  __shared__ float red[256];
  const int b = blockIdx.x, tid = threadIdx.x;
  const float* s = scores + (size_t)b * S_LEN;
  float* w = weights + (size_t)b * S_LEN;

  float m = -3.402823466e38f;
  for (int i = tid; i < S_LEN; i += 256) m = fmaxf(m, s[i]);
  red[tid] = m;
  __syncthreads();
  for (int off = 128; off > 0; off >>= 1) {
    if (tid < off) red[tid] = fmaxf(red[tid], red[tid + off]);
    __syncthreads();
  }
  const float M = red[0];
  __syncthreads();

  float sum = 0.0f;
  for (int i = tid; i < S_LEN; i += 256) {
    float e = __expf(s[i] - M);
    w[i] = e;
    sum += e;
  }
  red[tid] = sum;
  __syncthreads();
  for (int off = 128; off > 0; off >>= 1) {
    if (tid < off) red[tid] += red[tid + off];
    __syncthreads();
  }
  const float inv = 1.0f / red[0];
  for (int i = tid; i < S_LEN; i += 256) w[i] *= inv;
}

// ---------------------------------------------------------------------------
// Kernel 4: context partials. grid (chunk=16, b=16); each WG reduces 512
// s-rows for all 512 d (2 floats/thread). Deterministic (no global atomics).
// ---------------------------------------------------------------------------
__global__ __launch_bounds__(256) void ctx_partial_kernel(
    const float* __restrict__ enc, const float* __restrict__ weights,
    float* __restrict__ partials) {
  __shared__ float sw[S_LEN / CTX_CH];
  const int ch = blockIdx.x, b = blockIdx.y, tid = threadIdx.x;
  const int rows = S_LEN / CTX_CH;   // 512
  const float* wsrc = weights + (size_t)b * S_LEN + (size_t)ch * rows;
  for (int i = tid; i < rows; i += 256) sw[i] = wsrc[i];
  __syncthreads();

  const float* base =
      enc + ((size_t)b * S_LEN + (size_t)ch * rows) * D_DIM + tid * 2;
  float ax = 0.0f, ay = 0.0f;
  for (int i = 0; i < rows; ++i) {
    float wv = sw[i];
    const float2 e = *(const float2*)(base + (size_t)i * D_DIM);
    ax += wv * e.x;
    ay += wv * e.y;
  }
  float* outp = partials + ((size_t)(b * CTX_CH + ch)) * D_DIM + tid * 2;
  outp[0] = ax;
  outp[1] = ay;
}

// ---------------------------------------------------------------------------
// Kernel 5: reduce chunks -> context[b,d]. Writes every d_out element.
// ---------------------------------------------------------------------------
__global__ __launch_bounds__(256) void ctx_reduce_kernel(
    const float* __restrict__ partials, float* __restrict__ out) {
  int idx = blockIdx.x * 256 + threadIdx.x;   // [0, 8192)
  int b = idx >> 9;
  int d = idx & (D_DIM - 1);
  float s = 0.0f;
#pragma unroll
  for (int c = 0; c < CTX_CH; ++c)
    s += partials[((size_t)(b * CTX_CH + c)) * D_DIM + d];
  out[idx] = s;
}

// ---------------------------------------------------------------------------
extern "C" void kernel_launch(void* const* d_in, const int* in_sizes, int n_in,
                              void* d_out, int out_size, void* d_ws,
                              size_t ws_size, hipStream_t stream) {
  const float* enc    = (const float*)d_in[0];  // [B,S,D]
  const float* hidden = (const float*)d_in[1];  // [B,D]
  const float* W      = (const float*)d_in[2];  // [H,H]
  const float* b_attn = (const float*)d_in[3];  // [H]
  const float* v_w    = (const float*)d_in[4];  // [H]
  float* out = (float*)d_out;                   // [B,D] = 8192 f32

  float* ws       = (float*)d_ws;
  float* hbias    = ws;                    // 16384 floats
  __bf16* Wp      = (__bf16*)(ws + 16384); // 524288 bf16 = 262144 floats
  float* scores   = ws + 16384 + 262144;   // 131072 floats
  float* weights  = scores + 131072;       // 131072 floats
  float* partials = weights + 131072;      // 131072 floats (~2.6 MB total)

  hbias_kernel<<<64, 256, 0, stream>>>(hidden, W, b_attn, hbias);
  wpack_kernel<<<128, 256, 0, stream>>>(W, Wp);
  score_kernel<<<B_DIM * S_LEN / 16, 256, 0, stream>>>(enc, Wp, hbias, v_w,
                                                       scores);
  softmax_kernel<<<B_DIM, 256, 0, stream>>>(scores, weights);
  ctx_partial_kernel<<<dim3(CTX_CH, B_DIM), 256, 0, stream>>>(enc, weights,
                                                              partials);
  ctx_reduce_kernel<<<32, 256, 0, stream>>>(partials, out);
}
